// PSI_91190745629309
// MI455X (gfx1250) — matrix operand order repaired and checked by reading, not compiled, not run on hardware
//
#include <hip/hip_runtime.h>
#include <math.h>

#define BB 16
#define SS 2048
#define VV 21
#define EE 64
#define HH 128
#define LCONV 2046   // S - K + 1 for K=3 VALID; kept by SAME convs afterwards
#define KT 64        // keys per attention iteration
#define NIT (SS / KT)

typedef __attribute__((ext_vector_type(16))) __bf16 v16bf;
typedef __attribute__((ext_vector_type(8)))  float  v8f;

// ---- ISA §7.12.2 layout helpers (wave32) -----------------------------------
// A-matrix 16x32 bf16: lane&15 = M; this code's K = (lane>>4)*16 + slot (contiguous, not ISA table order).
__device__ __forceinline__ int a_kidx(int lane, int slot) {
    return (lane >> 4) * 16 + slot;
}
// B-matrix 32x16 bf16: N = lane&15, this code's K = (lane>>4)*16 + slot (same order as A).
// C/D f32 16x16: element r of v8f -> (m = r + 8*(lane>>4), n = lane&15).

// xor-butterfly max within each 16-lane row via v_permlane16_b32 (VALU, no LDS)
__device__ __forceinline__ float permmax(float v, unsigned lo, unsigned hi_) {
    int p = __builtin_amdgcn_permlane16(__float_as_int(v), __float_as_int(v),
                                        lo, hi_, true, false);
    return fmaxf(v, __int_as_float(p));
}
__device__ __forceinline__ float rowmax16(float v) {
    v = permmax(v, 0x67452301u, 0xEFCDAB89u);  // xor 1
    v = permmax(v, 0x54761032u, 0xDCFE98BAu);  // xor 2
    v = permmax(v, 0x32107654u, 0xBA98FEDCu);  // xor 4
    v = permmax(v, 0xFEDCBA98u, 0x76543210u);  // xor 8
    return v;
}

// ---- CDNA5 async global->LDS copy (ASYNCcnt) --------------------------------
// Low 32 bits of a flat shared-memory address are the LDS byte address
// (ISA 10.2).
__device__ __forceinline__ unsigned lds_addr32(const void* p) {
    return (unsigned)(uintptr_t)p;
}
__device__ __forceinline__ void async_b128(unsigned lds_a, const void* gaddr) {
    asm volatile("global_load_async_to_lds_b128 %0, %1, off"
                 :: "v"(lds_a), "v"(gaddr) : "memory");
}
__device__ __forceinline__ void wait_async0() {
    asm volatile("s_wait_asynccnt 0x0" ::: "memory");
}

// ---- weight f32 -> bf16, transposed (dst[n][k] = src[k][n], 128x128) --------
__global__ void PSI_tobf16T(const float* __restrict__ src, __bf16* __restrict__ dst) {
    int i = blockIdx.x * blockDim.x + threadIdx.x;
    if (i < HH * HH) {
        int k = i >> 7, n = i & 127;
        dst[n * HH + k] = (__bf16)src[i];
    }
}

// ---- embedding lookup + LN(E=64) + Linear E->H ------------------------------
__global__ __launch_bounds__(128) void PSI_embed(
    const float* __restrict__ x, const float* __restrict__ emb,
    const float* __restrict__ g0, const float* __restrict__ b0,
    const float* __restrict__ Wemb, const float* __restrict__ bemb,
    float* __restrict__ hf, __bf16* __restrict__ hb) {
    __shared__ float e[EE];
    __shared__ float stat[2];
    const int tok = blockIdx.x;
    const int tid = threadIdx.x;
    const float* xp = x + (size_t)tok * VV;
    if (tid < EE) {
        float a = 0.f;
        for (int v = 0; v < VV; ++v) a += xp[v] * emb[v * EE + tid];
        e[tid] = a;
    }
    __syncthreads();
    if (tid == 0) {
        float mu = 0.f;
        for (int i = 0; i < EE; ++i) mu += e[i];
        mu *= (1.f / EE);
        float var = 0.f;
        for (int i = 0; i < EE; ++i) { float d = e[i] - mu; var += d * d; }
        var *= (1.f / EE);
        stat[0] = mu; stat[1] = rsqrtf(var + 1e-5f);
    }
    __syncthreads();
    if (tid < EE) e[tid] = (e[tid] - stat[0]) * stat[1] * g0[tid] + b0[tid];
    __syncthreads();
    float a = bemb[tid];
    for (int k = 0; k < EE; ++k) a += e[k] * Wemb[k * HH + tid];
    hf[(size_t)tok * HH + tid] = a;
    hb[(size_t)tok * HH + tid] = (__bf16)a;
}

// ---- flash attention: softmax(h h^T) h * inv_scale, bf16 WMMA ---------------
// 8 waves/block, each wave one 16-row q-tile; 64-key tiles double-buffered in
// LDS via async global->LDS loads so the fetch overlaps the WMMA compute.
__global__ __launch_bounds__(256) void PSI_attn(
    const __bf16* __restrict__ hb, __bf16* __restrict__ vb) {
    __shared__ __bf16 kt[2][KT * HH];     // double-buffered key tile [key][h]
    __shared__ __bf16 ktT[HH * KT];       // transposed current tile  [h][key]
    __shared__ __bf16 pl[8][16 * KT];     // per-wave P exchange (C->A layout)
    const int lane = threadIdx.x & 31;
    const int wave = threadIdx.x >> 5;
    const int tile = blockIdx.x * 8 + wave;   // 2048 q-tiles total
    const int b  = tile >> 7;                 // 128 q-tiles per batch
    const int q0 = (tile & 127) << 4;
    const int m  = lane & 15;
    const int hi = lane >> 4;

    // Q tile, A layout, 4 chunks of K=32 over the H=128 contraction
    v16bf aq[4];
    const __bf16* qrow = hb + ((size_t)b * SS + q0 + m) * HH;
#pragma unroll
    for (int kc = 0; kc < 4; ++kc)
#pragma unroll
        for (int s = 0; s < 16; ++s)
            aq[kc][s] = qrow[kc * 32 + a_kidx(lane, s)];

    v16bf bones;   // all-ones B for row-sum WMMA
#pragma unroll
    for (int s = 0; s < 16; ++s) bones[s] = (__bf16)1.0f;

    v8f acc[8];
#pragma unroll
    for (int nt = 0; nt < 8; ++nt) acc[nt] = (v8f){};
    float mrun[8], lrun[8];
#pragma unroll
    for (int r = 0; r < 8; ++r) { mrun[r] = -1e30f; lrun[r] = 0.f; }

    const char* kbase = (const char*)(hb + (size_t)b * SS * HH);
    // prologue: async-load tile 0 into buffer 0 (16KB: 4 x b128 per thread)
    {
        unsigned la = lds_addr32(&kt[0][0]) + threadIdx.x * 16u;
        const char* ga = kbase + threadIdx.x * 16u;
#pragma unroll
        for (int i = 0; i < 4; ++i)
            async_b128(la + i * 4096u, ga + (size_t)i * 4096);
    }

    for (int it = 0; it < NIT; ++it) {
        const int cur = it & 1;
        wait_async0();       // this wave's share of tile `it` has landed
        __syncthreads();     // whole tile landed; prev iteration reads done

        if (it + 1 < NIT) {  // overlap next tile fetch with compute
            unsigned la = lds_addr32(&kt[cur ^ 1][0]) + threadIdx.x * 16u;
            const char* ga = kbase + (size_t)(it + 1) * KT * HH * 2 + threadIdx.x * 16u;
#pragma unroll
            for (int i = 0; i < 4; ++i)
                async_b128(la + i * 4096u, ga + (size_t)i * 4096);
        }

        {   // LDS->LDS transpose: ktT[h][key] = kt[cur][key][h]
#pragma unroll
            for (int blk = 0; blk < 4; ++blk) {
                int v4 = threadIdx.x * 4 + blk;
                int e = v4 * 8;                 // 8 contiguous bf16, one key row
                int k = e >> 7, h0 = e & 127;
                int4 d = ((const int4*)kt[cur])[v4];
                const __bf16* p = (const __bf16*)&d;
#pragma unroll
                for (int j = 0; j < 8; ++j) ktT[(h0 + j) * KT + k] = p[j];
            }
        }

        // scores[16q, 64key] as four 16x16 C tiles (reads row-major kt[cur])
        v8f c[4];
#pragma unroll
        for (int kn = 0; kn < 4; ++kn) {
            v8f cc = (v8f){};
#pragma unroll
            for (int kc = 0; kc < 4; ++kc) {
                v16bf bk;   // B[k=h, n=key]: contiguous over slot
#pragma unroll
                for (int s = 0; s < 16; ++s)
                    bk[s] = kt[cur][(kn * 16 + m) * HH + kc * 32 + hi * 16 + s];
                cc = __builtin_amdgcn_wmma_f32_16x16x32_bf16(
                        false, aq[kc], false, bk, (short)0, cc, false, false);
            }
            c[kn] = cc;
        }

        // online softmax: max via permlane butterflies, sum via ones-WMMA below
        float alpha[8];
#pragma unroll
        for (int r = 0; r < 8; ++r) {
            float t = fmaxf(fmaxf(c[0][r], c[1][r]), fmaxf(c[2][r], c[3][r]));
            t = rowmax16(t);
            float mn = fmaxf(mrun[r], t);
            alpha[r] = __expf(mrun[r] - mn);
            mrun[r] = mn;
#pragma unroll
            for (int kn = 0; kn < 4; ++kn) c[kn][r] = __expf(c[kn][r] - mn);
#pragma unroll
            for (int nt = 0; nt < 8; ++nt) acc[nt][r] *= alpha[r];
        }

        // P: C layout -> A layout via per-wave LDS
#pragma unroll
        for (int r = 0; r < 8; ++r)
#pragma unroll
            for (int kn = 0; kn < 4; ++kn)
                pl[wave][(r + 8 * hi) * KT + kn * 16 + m] = (__bf16)c[kn][r];
        __syncthreads();   // pl + ktT visible

        v16bf pa[2];
#pragma unroll
        for (int j = 0; j < 2; ++j)
#pragma unroll
            for (int s = 0; s < 16; ++s)
                pa[j][s] = pl[wave][m * KT + j * 32 + a_kidx(lane, s)];

        // row sums: P @ ones -> every C element of row r = rowsum(r)
        {
            v8f ls = (v8f){};
            ls = __builtin_amdgcn_wmma_f32_16x16x32_bf16(
                    false, pa[0], false, bones, (short)0, ls, false, false);
            ls = __builtin_amdgcn_wmma_f32_16x16x32_bf16(
                    false, pa[1], false, bones, (short)0, ls, false, false);
#pragma unroll
            for (int r = 0; r < 8; ++r) lrun[r] = lrun[r] * alpha[r] + ls[r];
        }

        // acc[16q, 128h] += P[16x64] @ V[64x128]  (reads transposed ktT)
#pragma unroll
        for (int nt = 0; nt < 8; ++nt)
#pragma unroll
            for (int j = 0; j < 2; ++j) {
                v16bf bv;   // B[k=key, n=h]: contiguous over slot in ktT
#pragma unroll
                for (int s = 0; s < 16; ++s)
                    bv[s] = ktT[(nt * 16 + m) * KT + j * 32 + hi * 16 + s];
                acc[nt] = __builtin_amdgcn_wmma_f32_16x16x32_bf16(
                            false, pa[j], false, bv, (short)0, acc[nt], false, false);
            }
    }

    const float inv_scale = 0.29730177875068026f;  // 1/128^0.25, applied post-softmax
#pragma unroll
    for (int nt = 0; nt < 8; ++nt)
#pragma unroll
        for (int r = 0; r < 8; ++r) {
            float v = acc[nt][r] * inv_scale / lrun[r];
            vb[((size_t)b * SS + q0 + r + 8 * hi) * HH + nt * 16 + m] = (__bf16)v;
        }
}

// ---- Y = act(X @ W + b), X:[M,128] bf16, Wt:[n][k] bf16 staged in LDS -------
__global__ __launch_bounds__(256) void PSI_gemm128(
    const __bf16* __restrict__ X, const __bf16* __restrict__ Wt,
    const float* __restrict__ bias, __bf16* __restrict__ Yb,
    float* __restrict__ Yf, int do_relu) {
    __shared__ __bf16 wl[HH * HH];   // transposed weights: wl[n*128 + k]
    const int lane = threadIdx.x & 31;
    const int wave = threadIdx.x >> 5;
    const int m  = lane & 15;
    const int hi = lane >> 4;
    {   // 32KB async weight stage: 8 x b128 per thread
        unsigned la = lds_addr32(wl) + threadIdx.x * 16u;
        const char* ga = (const char*)Wt + threadIdx.x * 16u;
#pragma unroll
        for (int i = 0; i < 8; ++i)
            async_b128(la + i * 4096u, ga + (size_t)i * 4096);
    }
    wait_async0();
    __syncthreads();

    const int tile = blockIdx.x * 8 + wave;
    const size_t row0 = (size_t)tile * 16;
    v16bf a[4];
    const __bf16* xrow = X + (row0 + m) * HH;
#pragma unroll
    for (int kc = 0; kc < 4; ++kc)
#pragma unroll
        for (int s = 0; s < 16; ++s)
            a[kc][s] = xrow[kc * 32 + a_kidx(lane, s)];

#pragma unroll
    for (int nt = 0; nt < 8; ++nt) {
        v8f c = (v8f){};
#pragma unroll
        for (int kc = 0; kc < 4; ++kc) {
            v16bf bw;   // B[k,n] = wl[n*128 + k]: contiguous over slot
#pragma unroll
            for (int s = 0; s < 16; ++s)
                bw[s] = wl[(nt * 16 + m) * HH + kc * 32 + hi * 16 + s];
            c = __builtin_amdgcn_wmma_f32_16x16x32_bf16(
                    false, a[kc], false, bw, (short)0, c, false, false);
        }
#pragma unroll
        for (int r = 0; r < 8; ++r) {
            int n = nt * 16 + m;
            float v = c[r] + bias[n];
            if (do_relu) v = fmaxf(v, 0.f);
            size_t idx = (row0 + r + 8 * hi) * HH + n;
            if (Yb) Yb[idx] = (__bf16)v;
            if (Yf) Yf[idx] = v;
        }
    }
}

// ---- h = LN(v + h) over H=128 -----------------------------------------------
__global__ __launch_bounds__(128) void PSI_residln(
    const float* __restrict__ t3, const float* __restrict__ gg,
    const float* __restrict__ bb, float* __restrict__ hf, __bf16* __restrict__ hb) {
    __shared__ float vsh[HH];
    __shared__ float stat[2];
    const size_t base = (size_t)blockIdx.x * HH;
    const int tid = threadIdx.x;
    float val = t3[base + tid] + hf[base + tid];
    vsh[tid] = val;
    __syncthreads();
    if (tid == 0) {
        float mu = 0.f;
        for (int i = 0; i < HH; ++i) mu += vsh[i];
        mu *= (1.f / HH);
        float var = 0.f;
        for (int i = 0; i < HH; ++i) { float d = vsh[i] - mu; var += d * d; }
        var *= (1.f / HH);
        stat[0] = mu; stat[1] = rsqrtf(var + 1e-5f);
    }
    __syncthreads();
    float o = (val - stat[0]) * stat[1] * gg[tid] + bb[tid];
    hf[base + tid] = o;
    hb[base + tid] = (__bf16)o;
}

// ---- conv1: [B,S,H] (channels-last h) -> relu conv H->32, K=3 VALID ----------
__global__ void PSI_conv1(const float* __restrict__ hf, const float* __restrict__ w,
                          const float* __restrict__ bias, float* __restrict__ y) {
    int idx = blockIdx.x * blockDim.x + threadIdx.x;
    if (idx >= BB * 32 * LCONV) return;
    int l = idx % LCONV;
    int o = (idx / LCONV) % 32;
    int b = idx / (LCONV * 32);
    float a = bias[o];
#pragma unroll
    for (int t = 0; t < 3; ++t) {
        const float* hp = hf + ((size_t)b * SS + l + t) * HH;
        const float* wp = w + o * HH * 3 + t;   // w[o][i][t]
        for (int i = 0; i < HH; ++i) a += hp[i] * wp[i * 3];
    }
    y[idx] = fmaxf(a, 0.f);
}

// ---- conv 32->32, K=3, SAME pad, relu ---------------------------------------
__global__ void PSI_conv_s(const float* __restrict__ xin, const float* __restrict__ w,
                           const float* __restrict__ bias, float* __restrict__ y) {
    int idx = blockIdx.x * blockDim.x + threadIdx.x;
    if (idx >= BB * 32 * LCONV) return;
    int l = idx % LCONV;
    int o = (idx / LCONV) % 32;
    int b = idx / (LCONV * 32);
    float a = bias[o];
#pragma unroll
    for (int t = 0; t < 3; ++t) {
        int ll = l + t - 1;
        if (ll < 0 || ll >= LCONV) continue;
        for (int i = 0; i < 32; ++i)
            a += xin[((size_t)b * 32 + i) * LCONV + ll] * w[o * 96 + i * 3 + t];
    }
    y[idx] = fmaxf(a, 0.f);
}

// ---- global max-pool + 32->32 relu -> 32->1 ---------------------------------
__global__ __launch_bounds__(64) void PSI_head(
    const float* __restrict__ y3, const float* __restrict__ m1w,
    const float* __restrict__ m1b, const float* __restrict__ m2w,
    const float* __restrict__ m2b, float* __restrict__ out) {
    __shared__ float g[32], hdn[32];
    const int b = blockIdx.x, tid = threadIdx.x;
    if (tid < 32) {
        float mx = -1e30f;
        const float* p = y3 + ((size_t)b * 32 + tid) * LCONV;
        for (int l = 0; l < LCONV; ++l) mx = fmaxf(mx, p[l]);
        g[tid] = mx;
    }
    __syncthreads();
    if (tid < 32) {
        float a = m1b[tid];
        for (int k = 0; k < 32; ++k) a += g[k] * m1w[k * 32 + tid];
        hdn[tid] = fmaxf(a, 0.f);
    }
    __syncthreads();
    if (tid == 0) {
        float a = m2b[0];
        for (int k = 0; k < 32; ++k) a += hdn[k] * m2w[k];
        out[b] = a;
    }
}

extern "C" void kernel_launch(void* const* d_in, const int* in_sizes, int n_in,
                              void* d_out, int out_size, void* d_ws, size_t ws_size,
                              hipStream_t stream) {
    (void)in_sizes; (void)n_in; (void)out_size; (void)ws_size;
    // inputs in setup_inputs() insertion order (depth-first)
    const float* x     = (const float*)d_in[0];
    const float* emb   = (const float*)d_in[1];
    const float* ln0_g = (const float*)d_in[2];
    const float* ln0_b = (const float*)d_in[3];
    const float* W_emb = (const float*)d_in[4];
    const float* b_emb = (const float*)d_in[5];
    const float *lW[3][3], *lb[3][3], *lg[3], *lbb[3];
    for (int l = 0; l < 3; ++l) {
        int base = 6 + l * 8;
        lW[l][0] = (const float*)d_in[base + 0];
        lb[l][0] = (const float*)d_in[base + 1];
        lW[l][1] = (const float*)d_in[base + 2];
        lb[l][1] = (const float*)d_in[base + 3];
        lW[l][2] = (const float*)d_in[base + 4];
        lb[l][2] = (const float*)d_in[base + 5];
        lg[l]    = (const float*)d_in[base + 6];
        lbb[l]   = (const float*)d_in[base + 7];
    }
    const float* c1w = (const float*)d_in[30];
    const float* c1b = (const float*)d_in[31];
    const float* c2w = (const float*)d_in[32];
    const float* c2b = (const float*)d_in[33];
    const float* c3w = (const float*)d_in[34];
    const float* c3b = (const float*)d_in[35];
    const float* m1w = (const float*)d_in[36];
    const float* m1b = (const float*)d_in[37];
    const float* m2w = (const float*)d_in[38];
    const float* m2b = (const float*)d_in[39];
    float* out = (float*)d_out;

    // workspace carve-out (~72 MB)
    char* ws = (char*)d_ws;
    size_t off = 0;
    auto alloc = [&](size_t bytes) -> char* {
        char* p = ws + off;
        off = (off + bytes + 255) & ~(size_t)255;
        return p;
    };
    const size_t NT = (size_t)BB * SS;          // 32768 tokens
    float*  hf   = (float*) alloc(NT * HH * 4);
    __bf16* hb   = (__bf16*)alloc(NT * HH * 2);
    __bf16* bufA = (__bf16*)alloc(NT * HH * 2);
    __bf16* bufB = (__bf16*)alloc(NT * HH * 2);
    float*  t3f  = (float*) alloc(NT * HH * 4);
    __bf16* wbf  = (__bf16*)alloc(9 * HH * HH * 2);
    float*  y1   = (float*) alloc((size_t)BB * 32 * LCONV * 4);
    float*  y2   = (float*) alloc((size_t)BB * 32 * LCONV * 4);
    float*  y3   = (float*) alloc((size_t)BB * 32 * LCONV * 4);

    // 1. weight conversion to bf16, transposed for contiguous B-operand reads
    for (int l = 0; l < 3; ++l)
        for (int j = 0; j < 3; ++j)
            PSI_tobf16T<<<(HH * HH + 255) / 256, 256, 0, stream>>>(
                lW[l][j], wbf + (size_t)(l * 3 + j) * HH * HH);

    // 2. embedding + LN + proj
    PSI_embed<<<NT, 128, 0, stream>>>(x, emb, ln0_g, ln0_b, W_emb, b_emb, hf, hb);

    // 3. transformer layers (flash-attention + WMMA MLPs)
    for (int l = 0; l < 3; ++l) {
        PSI_attn<<<256, 256, 0, stream>>>(hb, bufA);
        PSI_gemm128<<<256, 256, 0, stream>>>(bufA, wbf + (size_t)(l * 3 + 0) * HH * HH,
                                             lb[l][0], bufB, nullptr, 1);
        PSI_gemm128<<<256, 256, 0, stream>>>(bufB, wbf + (size_t)(l * 3 + 1) * HH * HH,
                                             lb[l][1], bufA, nullptr, 1);
        PSI_gemm128<<<256, 256, 0, stream>>>(bufA, wbf + (size_t)(l * 3 + 2) * HH * HH,
                                             lb[l][2], nullptr, t3f, 0);
        PSI_residln<<<NT, 128, 0, stream>>>(t3f, lg[l], lbb[l], hf, hb);
    }

    // 4. decoder head
    const int nconv = BB * 32 * LCONV;
    PSI_conv1 <<<(nconv + 255) / 256, 256, 0, stream>>>(hf, c1w, c1b, y1);
    PSI_conv_s<<<(nconv + 255) / 256, 256, 0, stream>>>(y1, c2w, c2b, y2);
    PSI_conv_s<<<(nconv + 255) / 256, 256, 0, stream>>>(y2, c3w, c3b, y3);
    PSI_head  <<<BB, 64, 0, stream>>>(y3, m1w, m1b, m2w, m2b, out);
}
